// TransLayer_87780541596185
// MI455X (gfx1250) — compile-verified, matrix operand order checked by reading, not run on hardware
//
#include <hip/hip_runtime.h>

// ---------------------------------------------------------------------------
// Nystromformer layer for MI455X (gfx1250): WMMA-f16 GEMM pipeline.
// Every GEMM computes A (row-major [M,K]) x B^T (row-major [N,K]) so both
// operand tiles stream through GLOBAL_LOAD_ASYNC_TO_LDS_B128 (ASYNCcnt),
// double-buffered in LDS, 8 v_wmma_f32_16x16x32_f16 per wave per k-step.
// ---------------------------------------------------------------------------

typedef _Float16 f16_t;
typedef __attribute__((ext_vector_type(8)))  _Float16 v8h;
typedef __attribute__((ext_vector_type(16))) _Float16 v16h;
typedef __attribute__((ext_vector_type(8)))  float    v8f;
typedef __attribute__((ext_vector_type(4)))  int      v4i;

#define B_      8
#define N_      4096
#define D_      512
#define H_      4
#define DH_     64
#define INNER_  256
#define M_      256
#define L_      16
#define BN_TOK  (B_ * N_)      // 32768 tokens

#define AS1 __attribute__((address_space(1)))
#define AS3 __attribute__((address_space(3)))

#if defined(__has_builtin)
#if __has_builtin(__builtin_amdgcn_global_load_async_to_lds_b128)
#define HAVE_ASYNC_LDS 1
#endif
#endif
#ifndef HAVE_ASYNC_LDS
#define HAVE_ASYNC_LDS 0
#endif

__device__ __forceinline__ void wait_asynccnt0() {
#if defined(__has_builtin) && __has_builtin(__builtin_amdgcn_s_wait_asynccnt)
    __builtin_amdgcn_s_wait_asynccnt(0);
#else
    asm volatile("s_wait_asynccnt 0" ::: "memory");
#endif
}

__device__ __forceinline__ void async_cp16(const f16_t* src, f16_t* dst_lds) {
#if HAVE_ASYNC_LDS
    __builtin_amdgcn_global_load_async_to_lds_b128(
        (AS1 v4i*)src, (AS3 v4i*)dst_lds, 0, 0);
#else
    *(v8h*)dst_lds = *(const v8h*)src;
#endif
}

__device__ __forceinline__ v16h ld_frag16(const f16_t* p) {
    v8h lo = *(const v8h*)p;
    v8h hi = *(const v8h*)(p + 8);
    v16h r;
#pragma unroll
    for (int i = 0; i < 8; ++i) { r[i] = lo[i]; r[i + 8] = hi[i]; }
    return r;
}

// ---------------------------------------------------------------------------
// Batched GEMM:  C = alpha * A * B^T + diag*I (+ bias[col]) (+ res)
//   A : f16 [M,K] lda;  B : f16 [N,K] ldb  (both row-major, K contiguous)
//   C32 (f32), C16 (f16), C16t (f16 transposed: C16t[col*ldct+row]) optional.
// Tile BM=128 x BN=64 x BK=32; 128 threads = 4 waves stacked on M;
// each wave: 32x64 = 2x4 WMMA fragments. LDS double-buffered, async-fed.
// All M mult of 128, N mult of 64, K mult of 32 in this problem.
// ---------------------------------------------------------------------------
__global__ __launch_bounds__(128)
void k_gemm(const f16_t* __restrict__ A, const f16_t* __restrict__ B,
            float* __restrict__ C32, f16_t* __restrict__ C16,
            f16_t* __restrict__ C16t,
            int M, int N, int K, int lda, int ldb, int ldc, int ldct,
            long sA, long sB, long sC,
            float alpha, float diag,
            const float* __restrict__ bias, const float* __restrict__ res)
{
    const int bz = blockIdx.z;
    A += (long)bz * sA;
    B += (long)bz * sB;
    if (C32)  C32  += (long)bz * sC;
    if (C16)  C16  += (long)bz * sC;
    if (C16t) C16t += (long)bz * sC;

    constexpr int BM = 128, BN = 64, BK = 32;
    constexpr int AST = BK + 8;            // 40 halves, conflict-padded
    __shared__ f16_t As[2 * BM * AST];     // 2 x 10 KB
    __shared__ f16_t Bs[2 * BN * AST];     // 2 x 5 KB

    const int tid  = threadIdx.x;
    const int wave = tid >> 5;             // 0..3 -> row band wave*32
    const int lane = tid & 31;
    const int tile_m = blockIdx.y * BM;
    const int tile_n = blockIdx.x * BN;
    const int l16   = lane & 15;
    const int khalf = (lane >> 4) << 4;    // 0 or 16

    v8f acc[2][4] = {};

    auto stage = [&](int buf, int k0) {
        f16_t* as = As + buf * (BM * AST);
        f16_t* bs = Bs + buf * (BN * AST);
        // A tile: 128x32 halves = 512 x 16B chunks, 4 per thread
#pragma unroll
        for (int i = 0; i < 4; ++i) {
            int idx = tid + i * 128;
            int r = idx >> 2, c8 = (idx & 3) << 3;
            async_cp16(A + (long)(tile_m + r) * lda + k0 + c8, &as[r * AST + c8]);
        }
        // B tile: 64x32 halves = 256 x 16B chunks, 2 per thread
#pragma unroll
        for (int i = 0; i < 2; ++i) {
            int idx = tid + i * 128;
            int r = idx >> 2, c8 = (idx & 3) << 3;
            async_cp16(B + (long)(tile_n + r) * ldb + k0 + c8, &bs[r * AST + c8]);
        }
    };

    auto compute = [&](int buf) {
        const f16_t* as = As + buf * (BM * AST);
        const f16_t* bs = Bs + buf * (BN * AST);
        v16h af[2], bf[4];
#pragma unroll
        for (int t = 0; t < 2; ++t)
            af[t] = ld_frag16(&as[(wave * 32 + t * 16 + l16) * AST + khalf]);
#pragma unroll
        for (int t = 0; t < 4; ++t)
            bf[t] = ld_frag16(&bs[(t * 16 + l16) * AST + khalf]);
#pragma unroll
        for (int tm = 0; tm < 2; ++tm)
#pragma unroll
            for (int tn = 0; tn < 4; ++tn)
                acc[tm][tn] = __builtin_amdgcn_wmma_f32_16x16x32_f16(
                    false, af[tm], false, bf[tn],
                    (short)0, acc[tm][tn], false, false);
    };

#if HAVE_ASYNC_LDS
    int buf = 0;
    stage(0, 0);
    wait_asynccnt0();
    __syncthreads();
    for (int k0 = 0; k0 < K; k0 += BK) {
        if (k0 + BK < K) stage(buf ^ 1, k0 + BK);   // async prefetch next tile
        compute(buf);
        wait_asynccnt0();
        __syncthreads();
        buf ^= 1;
    }
#else
    for (int k0 = 0; k0 < K; k0 += BK) {
        stage(0, k0);
        __syncthreads();
        compute(0);
        __syncthreads();
    }
#endif

    // Epilogue: C layout VGPR r -> M = r (lanes 0-15) / r+8 (lanes 16-31)
    const int radd = (lane >> 4) * 8;
#pragma unroll
    for (int tm = 0; tm < 2; ++tm) {
#pragma unroll
        for (int tn = 0; tn < 4; ++tn) {
            int gcol = tile_n + tn * 16 + l16;
#pragma unroll
            for (int r = 0; r < 8; ++r) {
                int grow = tile_m + wave * 32 + tm * 16 + r + radd;
                float v = alpha * acc[tm][tn][r];
                if (grow == gcol) v += diag;
                if (bias) v += bias[gcol];
                if (res)  v += res[(long)grow * ldc + gcol];
                if (C32)  C32 [(long)grow * ldc  + gcol] = v;
                if (C16)  C16 [(long)grow * ldc  + gcol] = (f16_t)v;
                if (C16t) C16t[(long)gcol * ldct + grow] = (f16_t)v;
            }
        }
    }
}

// --------------------------- LayerNorm -> f16 -------------------------------
__global__ __launch_bounds__(128)
void k_layernorm(const float* __restrict__ x, const float* __restrict__ g,
                 const float* __restrict__ b, f16_t* __restrict__ xn)
{
    long row = blockIdx.x;
    const float* xp = x + row * D_;
    int tid = threadIdx.x;
    float v[4]; float s = 0.f, s2 = 0.f;
#pragma unroll
    for (int i = 0; i < 4; ++i) {
        float t = xp[tid + i * 128];
        v[i] = t; s += t; s2 += t * t;
    }
    __shared__ float rs[128], rq[128];
    rs[tid] = s; rq[tid] = s2; __syncthreads();
    for (int k = 64; k > 0; k >>= 1) {
        if (tid < k) { rs[tid] += rs[tid + k]; rq[tid] += rq[tid + k]; }
        __syncthreads();
    }
    float mu   = rs[0] * (1.f / D_);
    float var  = rq[0] * (1.f / D_) - mu * mu;
    float rstd = rsqrtf(var + 1e-5f);
    f16_t* op = xn + row * D_;
#pragma unroll
    for (int i = 0; i < 4; ++i) {
        int c = tid + i * 128;
        op[c] = (f16_t)((v[i] - mu) * rstd * g[c] + b[c]);
    }
}

// --------- f32 [K,N] -> f16 transposed [N,K], tiled via LDS ----------------
__global__ __launch_bounds__(256)
void k_cvt_tr(const float* __restrict__ in, f16_t* __restrict__ outT,
              int K, int N)
{
    __shared__ float t[64][65];
    int c = threadIdx.x & 63, r4 = threadIdx.x >> 6;
    int k0 = blockIdx.y * 64, n0 = blockIdx.x * 64;
#pragma unroll
    for (int p = 0; p < 16; ++p) {
        int k = r4 + p * 4;
        t[k][c] = in[(long)(k0 + k) * N + n0 + c];
    }
    __syncthreads();
#pragma unroll
    for (int p = 0; p < 16; ++p) {
        int n = r4 + p * 4;
        outT[(long)(n0 + n) * K + k0 + c] = (f16_t)t[c][n];
    }
}

// -- outT[j,i] = sgn*in[i,j]*sc + delta*(i==j); 256x256 per batch, LDS-tiled -
__global__ __launch_bounds__(256)
void k_tr_scale_diag(const float* __restrict__ in, f16_t* __restrict__ outT,
                     float sgn, float delta, const float* __restrict__ scal)
{
    const int bh = blockIdx.z;
    const int ti = (blockIdx.x >> 2) * 64;   // input row tile
    const int tj = (blockIdx.x & 3) * 64;    // input col tile
    in   += (long)bh * M_ * M_;
    outT += (long)bh * M_ * M_;
    __shared__ float t[64][65];
    int c = threadIdx.x & 63, r4 = threadIdx.x >> 6;
    float sc = sgn * (scal ? 1.f / (scal[0] * scal[1]) : 1.f);
#pragma unroll
    for (int p = 0; p < 16; ++p) {
        int i = r4 + p * 4;
        t[i][c] = in[(long)(ti + i) * M_ + tj + c];
    }
    __syncthreads();
#pragma unroll
    for (int p = 0; p < 16; ++p) {
        int j = r4 + p * 4;
        int gi = ti + c, gj = tj + j;
        float v = t[c][j] * sc + ((gi == gj) ? delta : 0.f);
        outT[(long)gj * M_ + gi] = (f16_t)v;
    }
}

// ------------- linear f32 -> f16 with optional 1/(s0*s1) scale --------------
__global__ void k_cvt_scale(const float* __restrict__ in, f16_t* __restrict__ out,
                            const float* __restrict__ scal, long n)
{
    long i = (long)blockIdx.x * 256 + threadIdx.x;
    if (i >= n) return;
    float sc = scal ? 1.f / (scal[0] * scal[1]) : 1.f;
    out[i] = (f16_t)(in[i] * sc);
}

// ------ split QKV, scale q, q/k -> [B,h,N,dh] f16, v -> [B,h,dh,N] f16 ------
__global__ void k_split_qkv(const float* __restrict__ qkv,
                            f16_t* __restrict__ q, f16_t* __restrict__ k,
                            f16_t* __restrict__ vT)
{
    long idx = (long)blockIdx.x * 256 + threadIdx.x;   // over 32768*256
    if (idx >= (long)BN_TOK * INNER_) return;
    long t = idx >> 8;           // token
    int  c = (int)(idx & 255);   // h*64+d
    int  h = c >> 6, d = c & 63;
    int  b = (int)(t >> 12), n = (int)(t & (N_ - 1));
    int  bh = b * H_ + h;
    long dst = ((long)bh * N_ + n) * DH_ + d;
    const float* p = qkv + t * (3 * INNER_);
    q[dst] = (f16_t)(p[c] * 0.125f);           // dh^-0.5
    k[dst] = (f16_t)p[INNER_ + c];
    vT[((long)bh * DH_ + d) * N_ + n] = (f16_t)p[2 * INNER_ + c];
}

// --------------------------- landmark means --------------------------------
__global__ void k_landmarks(const f16_t* __restrict__ q, const f16_t* __restrict__ k,
                            f16_t* __restrict__ ql, f16_t* __restrict__ kl)
{
    long idx = (long)blockIdx.x * 256 + threadIdx.x;   // over 32*256*64
    if (idx >= (long)B_ * H_ * M_ * DH_) return;
    long bh  = idx / (M_ * DH_);
    int  rem = (int)(idx - bh * (M_ * DH_));
    int  j = rem >> 6, d = rem & 63;
    long base = ((long)bh * N_ + (long)j * L_) * DH_ + d;
    float sq = 0.f, sk = 0.f;
#pragma unroll
    for (int i = 0; i < L_; ++i) {
        sq += (float)q[base + (long)i * DH_];
        sk += (float)k[base + (long)i * DH_];
    }
    ql[idx] = (f16_t)(sq * (1.f / L_));
    kl[idx] = (f16_t)(sk * (1.f / L_));
}

// --------------------------- row softmax -----------------------------------
__global__ __launch_bounds__(256)
void k_softmax(const float* __restrict__ in, float* __restrict__ o32,
               f16_t* __restrict__ o16, int Lrow)
{
    long row = blockIdx.x;
    const float* rp = in + row * (long)Lrow;
    int tid = threadIdx.x;
    int nc = Lrow >> 8;                 // 1 or 16
    float vals[16];
    float m = -1e30f;
    for (int c = 0; c < nc; ++c) {
        float v = rp[c * 256 + tid];
        vals[c] = v; m = fmaxf(m, v);
    }
    __shared__ float red[256];
    red[tid] = m; __syncthreads();
    for (int s = 128; s > 0; s >>= 1) {
        if (tid < s) red[tid] = fmaxf(red[tid], red[tid + s]);
        __syncthreads();
    }
    m = red[0]; __syncthreads();
    float sum = 0.f;
    for (int c = 0; c < nc; ++c) { float e = __expf(vals[c] - m); vals[c] = e; sum += e; }
    red[tid] = sum; __syncthreads();
    for (int s = 128; s > 0; s >>= 1) {
        if (tid < s) red[tid] += red[tid + s];
        __syncthreads();
    }
    float inv = 1.f / red[0];
    for (int c = 0; c < nc; ++c) {
        float o = vals[c] * inv;
        long off = row * (long)Lrow + c * 256 + tid;
        if (o32) o32[off] = o;
        if (o16) o16[off] = (f16_t)o;
    }
}

// ------------------ pinv init: global max row/col abs-sums ------------------
__global__ void k_zero2(float* s) { if (threadIdx.x < 2) s[threadIdx.x] = 0.f; }

__global__ __launch_bounds__(256)
void k_pinv_maxes(const float* __restrict__ a2, float* __restrict__ scal)
{
    int bh = blockIdx.x, tid = threadIdx.x;
    const float* p = a2 + (long)bh * M_ * M_;
    float rs = 0.f, cs = 0.f;
    for (int j = 0; j < M_; ++j) {
        rs += fabsf(p[(long)tid * M_ + j]);   // col = abs.sum(-1)
        cs += fabsf(p[(long)j * M_ + tid]);   // row = abs.sum(-2)
    }
    __shared__ float red[256];
    red[tid] = rs; __syncthreads();
    for (int s = 128; s > 0; s >>= 1) { if (tid < s) red[tid] = fmaxf(red[tid], red[tid + s]); __syncthreads(); }
    if (tid == 0) atomicMax((int*)&scal[0], __float_as_int(red[0]));
    __syncthreads();
    red[tid] = cs; __syncthreads();
    for (int s = 128; s > 0; s >>= 1) { if (tid < s) red[tid] = fmaxf(red[tid], red[tid + s]); __syncthreads(); }
    if (tid == 0) atomicMax((int*)&scal[1], __float_as_int(red[0]));
}

// ---------------- depthwise conv(33) residual + head merge ------------------
// vT layout [B,h,dh,N] -> taps are contiguous along N.
__global__ void k_conv_merge(const float* __restrict__ oat, const f16_t* __restrict__ vT,
                             const float* __restrict__ wres, f16_t* __restrict__ out)
{
    long idx = (long)blockIdx.x * 256 + threadIdx.x;   // over 32*4096*64
    if (idx >= (long)B_ * H_ * N_ * DH_) return;
    int  d  = (int)(idx & 63);
    long r  = idx >> 6;
    int  n  = (int)(r & (N_ - 1));
    int  bh = (int)(r >> 12);
    int  h  = bh & (H_ - 1);
    int  b  = bh >> 2;
    float acc = oat[idx];
    const f16_t* vb = vT + ((long)bh * DH_ + d) * N_;
#pragma unroll 1
    for (int j = 0; j < 33; ++j) {
        int nn = n + j - 16;
        if (nn >= 0 && nn < N_) acc += wres[h * 33 + j] * (float)vb[nn];
    }
    out[(((long)b * N_ + n) * INNER_) + h * DH_ + d] = (f16_t)acc;
}

// ---------------------------------------------------------------------------
extern "C" void kernel_launch(void* const* d_in, const int* in_sizes, int n_in,
                              void* d_out, int out_size, void* d_ws, size_t ws_size,
                              hipStream_t stream)
{
    const float* x     = (const float*)d_in[0];
    const float* ln_g  = (const float*)d_in[1];
    const float* ln_b  = (const float*)d_in[2];
    const float* w_qkv = (const float*)d_in[3];
    const float* w_out = (const float*)d_in[4];
    const float* b_out = (const float*)d_in[5];
    const float* w_res = (const float*)d_in[6];
    float* out = (float*)d_out;

    char* w = (char*)d_ws;
    auto alloc = [&](size_t bytes) -> char* {
        char* p = w; w += (bytes + 255) & ~(size_t)255; return p;
    };
    f16_t* xn16    = (f16_t*)alloc((size_t)BN_TOK * D_ * 2);           // 33.5 MB
    f16_t* wqkvT16 = (f16_t*)alloc((size_t)D_ * 3 * INNER_ * 2);       // [768,512]
    f16_t* woutT16 = (f16_t*)alloc((size_t)INNER_ * D_ * 2);           // [512,256]
    char*  qkvreg  = alloc((size_t)BN_TOK * 3 * INNER_ * 4);           // 100.7 MB
    float* qkv32   = (float*)qkvreg;
    f16_t* q16     = (f16_t*)alloc((size_t)B_ * H_ * N_ * DH_ * 2);
    f16_t* k16     = (f16_t*)alloc((size_t)B_ * H_ * N_ * DH_ * 2);
    f16_t* vT16    = (f16_t*)alloc((size_t)B_ * H_ * N_ * DH_ * 2);    // [bh,dh,N]
    f16_t* ql16    = (f16_t*)alloc((size_t)B_ * H_ * M_ * DH_ * 2);
    f16_t* kl16    = (f16_t*)alloc((size_t)B_ * H_ * M_ * DH_ * 2);
    float* sim32   = (float*)alloc((size_t)B_ * H_ * N_ * M_ * 4);     // 134 MB
    f16_t* a1_16   = (f16_t*)alloc((size_t)B_ * H_ * N_ * M_ * 2);
    f16_t* a3_16   = (f16_t*)alloc((size_t)B_ * H_ * M_ * N_ * 2);
    float* a2_32   = (float*)alloc((size_t)B_ * H_ * M_ * M_ * 4);
    f16_t* a2_16   = (f16_t*)alloc((size_t)B_ * H_ * M_ * M_ * 2);
    f16_t* zA16    = (f16_t*)alloc((size_t)B_ * H_ * M_ * M_ * 2);
    f16_t* zAT16   = (f16_t*)alloc((size_t)B_ * H_ * M_ * M_ * 2);
    f16_t* zB16    = (f16_t*)alloc((size_t)B_ * H_ * M_ * M_ * 2);
    f16_t* zBT16   = (f16_t*)alloc((size_t)B_ * H_ * M_ * M_ * 2);
    float* xz32    = (float*)alloc((size_t)B_ * H_ * M_ * M_ * 4);
    f16_t* xz16    = (f16_t*)alloc((size_t)B_ * H_ * M_ * M_ * 2);
    float* u32     = (float*)alloc((size_t)B_ * H_ * M_ * M_ * 4);
    f16_t* uT16    = (f16_t*)alloc((size_t)B_ * H_ * M_ * M_ * 2);
    f16_t* tavT16  = (f16_t*)alloc((size_t)B_ * H_ * M_ * DH_ * 2);    // [bh,dh,m]
    float* scal    = (float*)alloc(256);
    // aliases (producers retired before reuse):
    f16_t* a1z16   = (f16_t*)qkvreg;                                   // 67.1 MB
    float* oattn32 = (float*)(qkvreg + (size_t)B_ * H_ * N_ * M_ * 2);
    f16_t* attn16  = (f16_t*)xn16;                                     // 16.8 MB

    const long sQ  = (long)N_ * DH_;    // per-(b,h) q/k/vT stride
    const long sL  = (long)M_ * DH_;    // landmark / tavT stride
    const long sNM = (long)N_ * M_;     // sim1/a1/a3 stride
    const long sMM = (long)M_ * M_;

    // 1) weights -> f16 transposed, LayerNorm -> f16
    k_cvt_tr<<<dim3(12, 8), 256, 0, stream>>>(w_qkv, wqkvT16, D_, 3 * INNER_);
    k_cvt_tr<<<dim3(8, 4), 256, 0, stream>>>(w_out, woutT16, INNER_, D_);
    k_layernorm<<<dim3(BN_TOK), 128, 0, stream>>>(x, ln_g, ln_b, xn16);

    // 2) QKV projection: [32768,512] x [768,512]^T
    k_gemm<<<dim3(12, 256, 1), 128, 0, stream>>>(xn16, wqkvT16, qkv32, nullptr, nullptr,
        BN_TOK, 3 * INNER_, D_, D_, D_, 3 * INNER_, 0,
        0, 0, 0, 1.f, 0.f, nullptr, nullptr);
    k_split_qkv<<<dim3(BN_TOK), 256, 0, stream>>>(qkv32, q16, k16, vT16);
    k_landmarks<<<dim3((B_ * H_ * M_ * DH_) / 256), 256, 0, stream>>>(q16, k16, ql16, kl16);

    // 3) sim1 = q . kl^T  -> softmax -> a1 (f16)
    k_gemm<<<dim3(4, 32, 32), 128, 0, stream>>>(q16, kl16, sim32, nullptr, nullptr,
        N_, M_, DH_, DH_, DH_, M_, 0, sQ, sL, sNM, 1.f, 0.f, nullptr, nullptr);
    k_softmax<<<dim3(B_ * H_ * N_), 256, 0, stream>>>(sim32, nullptr, a1_16, M_);

    // 4) sim2 = ql . kl^T -> softmax -> a2 (f32 + f16)
    k_gemm<<<dim3(4, 2, 32), 128, 0, stream>>>(ql16, kl16, a2_32, nullptr, nullptr,
        M_, M_, DH_, DH_, DH_, M_, 0, sL, sL, sMM, 1.f, 0.f, nullptr, nullptr);
    k_softmax<<<dim3(B_ * H_ * M_), 256, 0, stream>>>(a2_32, a2_32, a2_16, M_);

    // 5) sim3 = ql . k^T  -> softmax -> a3 (f16)
    k_gemm<<<dim3(64, 2, 32), 128, 0, stream>>>(ql16, k16, sim32, nullptr, nullptr,
        M_, N_, DH_, DH_, DH_, N_, 0, sL, sQ, sNM, 1.f, 0.f, nullptr, nullptr);
    k_softmax<<<dim3(B_ * H_ * M_), 256, 0, stream>>>(sim32, nullptr, a3_16, N_);

    // 6) Moore-Penrose pinv of a2 (6 Newton-Schulz iterations, f16 GEMMs)
    k_zero2<<<dim3(1), 32, 0, stream>>>(scal);
    k_pinv_maxes<<<dim3(B_ * H_), 256, 0, stream>>>(a2_32, scal);
    // z0 = a2^T / s (transpose), z0^T = a2 / s (linear)
    k_tr_scale_diag<<<dim3(16, 1, 32), 256, 0, stream>>>(a2_32, zA16, 1.f, 0.f, scal);
    k_cvt_scale<<<dim3((B_ * H_ * M_ * M_) / 256), 256, 0, stream>>>(a2_32, zAT16, scal, (long)B_ * H_ * M_ * M_);

    f16_t *zc = zA16, *zcT = zAT16, *zn = zB16, *znT = zBT16;
    const dim3 gmm(4, 2, 32);
    const dim3 gtr(16, 1, 32);
    for (int it = 0; it < 6; ++it) {
        // xz = a2 @ z  (B = z^T)
        k_gemm<<<gmm, 128, 0, stream>>>(a2_16, zcT, xz32, xz16, nullptr,
            M_, M_, M_, M_, M_, M_, 0, sMM, sMM, sMM, 1.f, 0.f, nullptr, nullptr);
        k_tr_scale_diag<<<gtr, 256, 0, stream>>>(xz32, uT16, -1.f, 7.f, nullptr);   // u1^T
        k_gemm<<<gmm, 128, 0, stream>>>(xz16, uT16, u32, nullptr, nullptr,          // u2
            M_, M_, M_, M_, M_, M_, 0, sMM, sMM, sMM, 1.f, 0.f, nullptr, nullptr);
        k_tr_scale_diag<<<gtr, 256, 0, stream>>>(u32, uT16, -1.f, 15.f, nullptr);   // u3^T
        k_gemm<<<gmm, 128, 0, stream>>>(xz16, uT16, u32, nullptr, nullptr,          // u4
            M_, M_, M_, M_, M_, M_, 0, sMM, sMM, sMM, 1.f, 0.f, nullptr, nullptr);
        k_tr_scale_diag<<<gtr, 256, 0, stream>>>(u32, uT16, -1.f, 13.f, nullptr);   // u5^T
        // z_new = 0.25 * z @ u5 -> both layouts from one GEMM
        k_gemm<<<gmm, 128, 0, stream>>>(zc, uT16, nullptr, zn, znT,
            M_, M_, M_, M_, M_, M_, M_, sMM, sMM, sMM, 0.25f, 0.f, nullptr, nullptr);
        f16_t* t;
        t = zc;  zc  = zn;  zn  = t;
        t = zcT; zcT = znT; znT = t;     // zc/zcT now hold latest z
    }

    // 7) tav^T = (a3 @ v)^T : A=a3 [256,4096], B=v^T [64,4096] -> C16t [64,256]
    k_gemm<<<dim3(1, 2, 32), 128, 0, stream>>>(a3_16, vT16, nullptr, nullptr, tavT16,
        M_, DH_, N_, N_, N_, DH_, M_, (long)M_ * N_, sQ, sL, 1.f, 0.f, nullptr, nullptr);
    // 8) a1z = a1 @ a2inv : B = zcT [256,256]
    k_gemm<<<dim3(4, 32, 32), 128, 0, stream>>>(a1_16, zcT, nullptr, a1z16, nullptr,
        N_, M_, M_, M_, M_, M_, 0, sNM, sMM, sNM, 1.f, 0.f, nullptr, nullptr);
    // 9) out_attn = a1z @ tav : B = tavT [64,256]
    k_gemm<<<dim3(1, 32, 32), 128, 0, stream>>>(a1z16, tavT16, oattn32, nullptr, nullptr,
        N_, DH_, M_, M_, M_, DH_, 0, sNM, sL, sQ, 1.f, 0.f, nullptr, nullptr);

    // 10) + depthwise conv(33) residual, merge heads -> [B*N, 256] f16
    k_conv_merge<<<dim3((B_ * H_ * N_ * DH_) / 256), 256, 0, stream>>>(oattn32, vT16, w_res, attn16);

    // 11) final: out = x + attn @ w_out + b_out   (fused epilogue, B = woutT)
    k_gemm<<<dim3(8, 256, 1), 128, 0, stream>>>(attn16, woutT16, out, nullptr, nullptr,
        BN_TOK, D_, INNER_, INNER_, INNER_, D_, 0, 0, 0, 0, 1.f, 0.f, b_out, x);

    (void)in_sizes; (void)n_in; (void)out_size; (void)ws_size;
}
